// NLB_41025527611419
// MI455X (gfx1250) — matrix operand order repaired and checked by reading, not compiled, not run on hardware
//
#include <hip/hip_runtime.h>

typedef __attribute__((ext_vector_type(16))) __bf16       v16bf;
typedef __attribute__((ext_vector_type(8)))  float        v8f;
typedef __attribute__((ext_vector_type(4)))  unsigned int v4u;

#define BATCH 8
#define C_IN  256
#define C_INT 128
#define NPIX  4096
#define EPS   1e-5f

#define KT   64    // attention key tile
#define LDC  136   // row stride for [*][128] c-major tiles (272B, 16B-aligned, 68 dw)
#define LDK  72    // row stride for transposed [*][64] tiles (144B, 16B-aligned, 36 dw)

__device__ __forceinline__ v8f wmma_bf16(v16bf a, v16bf b, v8f c) {
  return __builtin_amdgcn_wmma_f32_16x16x32_bf16(false, a, false, b, (short)0, c,
                                                 false, false);
}

// ---- vectorized fragment loads (16B-aligned contiguous chunks) -------------
union pk16 { v4u u[2]; v16bf b; };
union pk8  { v4u u; __bf16 h[8]; };
union pk2  { unsigned u; __bf16 h[2]; };

// A-fragment (16x32 16-bit). Caller folds half*8 into base. Lane m = lane&15.
//   elems a[0..7]  = base[0..7]   (K 0..7  of this half)
//   elems a[8..15] = base[16..23] (K 16..23 of this half)
__device__ __forceinline__ v16bf load_a16(const __bf16* __restrict__ p) {
  pk16 c;
  c.u[0] = *(const v4u*)p;
  c.u[1] = *(const v4u*)(p + 16);
  return c.b;
}

// B-fragment (32x16 16-bit) where logical B[k][n] is stored transposed as
// base[n*ld + k]: per lane base[0..7] and base[16..23] (caller folds half*8 into &T[n][k0]).
__device__ __forceinline__ v16bf load_bt16(const __bf16* __restrict__ p) {
  pk16 c;
  c.u[0] = *(const v4u*)p;
  c.u[1] = *(const v4u*)(p + 16);
  return c.b;
}

// A-fragment from fp32 global rows (weights; L2-resident, compiler merges)
__device__ __forceinline__ v16bf load_a_f32(const float* __restrict__ row) {
  v16bf a;
#pragma unroll
  for (int i = 0; i < 8; ++i) {
    int k = ((i & 4) << 2) + 2 * (i & 3);
    a[2 * i]     = (__bf16)row[k];
    a[2 * i + 1] = (__bf16)row[k + 1];
  }
  return a;
}

// ---- CDNA5 async global->LDS copy (16B per lane) ---------------------------
__device__ __forceinline__ void async_copy16(unsigned lds_addr,
                                             const void* gptr) {
  asm volatile("global_load_async_to_lds_b128 %0, %1, off"
               :: "v"(lds_addr), "v"(gptr)
               : "memory");
}
__device__ __forceinline__ void async_wait0() {
  asm volatile("s_wait_asynccnt 0x0" ::: "memory");
}

// ---------------------------------------------------------------------------
// Kernel 1: fused BatchNorm + 1x1-conv projection, transposed bf16 output
//   outT[b][n][ci] = sum_c Wp[ci][c] * (x[b][c][n]*scale[c]+shift[c]) + bp[ci]
// ---------------------------------------------------------------------------
__global__ __launch_bounds__(256) void proj_kernel(
    const float* __restrict__ x, const float* __restrict__ gamma,
    const float* __restrict__ beta, const float* __restrict__ mean,
    const float* __restrict__ var, const float* __restrict__ Wp,
    const float* __restrict__ bp, __bf16* __restrict__ outT) {
  __shared__ __bf16 xs_t[128][LDK];  // BN(x) tile, stored [col][k]

  const int b    = blockIdx.y;
  const int col0 = blockIdx.x * 128;
  const int tid  = threadIdx.x;
  const int lane = tid & 31;
  const int wave = tid >> 5;
  const int half = lane >> 4;
  const int l16  = lane & 15;

  v8f acc[8];
#pragma unroll
  for (int i = 0; i < 8; ++i)
#pragma unroll
    for (int r = 0; r < 8; ++r) acc[i][r] = 0.0f;

  const float* xb = x + (size_t)b * C_IN * NPIX;

  for (int kc = 0; kc < C_IN; kc += 64) {
    // stage BN(x) (64 k x 128 cols) transposed; k-pairs packed into b32 stores
    for (int r = (tid >> 5) * 2; r < 64; r += 16) {
      const int ka = kc + r, kb = kc + r + 1;
      const float sa = gamma[ka] * rsqrtf(var[ka] + EPS);
      const float ha = beta[ka] - mean[ka] * sa;
      const float sb = gamma[kb] * rsqrtf(var[kb] + EPS);
      const float hb = beta[kb] - mean[kb] * sb;
      const int c4 = (tid & 31) * 4;
      const float4 va = *(const float4*)(xb + (size_t)ka * NPIX + col0 + c4);
      const float4 vb = *(const float4*)(xb + (size_t)kb * NPIX + col0 + c4);
      pk2 p;
      p.h[0] = (__bf16)(va.x * sa + ha); p.h[1] = (__bf16)(vb.x * sb + hb);
      *(unsigned*)&xs_t[c4 + 0][r] = p.u;
      p.h[0] = (__bf16)(va.y * sa + ha); p.h[1] = (__bf16)(vb.y * sb + hb);
      *(unsigned*)&xs_t[c4 + 1][r] = p.u;
      p.h[0] = (__bf16)(va.z * sa + ha); p.h[1] = (__bf16)(vb.z * sb + hb);
      *(unsigned*)&xs_t[c4 + 2][r] = p.u;
      p.h[0] = (__bf16)(va.w * sa + ha); p.h[1] = (__bf16)(vb.w * sb + hb);
      *(unsigned*)&xs_t[c4 + 3][r] = p.u;
    }
    __syncthreads();

#pragma unroll
    for (int kk = 0; kk < 64; kk += 32) {
      const int m = wave * 16 + l16;
      v16bf af = load_a_f32(Wp + (size_t)m * C_IN + kc + kk + half * 8);
#pragma unroll
      for (int nt = 0; nt < 8; ++nt) {
        v16bf bf = load_bt16(&xs_t[nt * 16 + l16][kk + half * 8]);
        acc[nt]  = wmma_bf16(af, bf, acc[nt]);
      }
    }
    __syncthreads();
  }

  // bias + packed 16B transposed store: outT[b][n][m0..m0+7]
  const int m0 = wave * 16 + 8 * half;
#pragma unroll
  for (int nt = 0; nt < 8; ++nt) {
    const int n  = col0 + nt * 16 + l16;
    __bf16* orow = outT + ((size_t)b * NPIX + n) * C_INT;
    pk8 pk;
#pragma unroll
    for (int r = 0; r < 8; ++r) pk.h[r] = (__bf16)(acc[nt][r] + bp[m0 + r]);
    *(v4u*)(orow + m0) = pk.u;
  }
}

// ---------------------------------------------------------------------------
// Kernel 2: flash attention.  qT/kT/gT: (B,N,Ci) bf16.  yT out: (B,N,Ci) bf16.
// ---------------------------------------------------------------------------
__global__ __launch_bounds__(256) void attn_kernel(
    const __bf16* __restrict__ qT, const __bf16* __restrict__ kT,
    const __bf16* __restrict__ gT, __bf16* __restrict__ yT) {
  __shared__ __bf16 kt[KT][LDC];      // key tile [key][c]   (async-copied)
  __shared__ __bf16 gt_t[128][LDK];   // g tile   [c][key]   (transposed)
  __shared__ __bf16 ps[8][16][LDK];   // per-wave P tile [qrow][key]

  const int b    = blockIdx.y;
  const int q0   = blockIdx.x * 128;
  const int tid  = threadIdx.x;
  const int lane = tid & 31;
  const int wave = tid >> 5;
  const int half = lane >> 4;
  const int l16  = lane & 15;

  const unsigned kt_base = (unsigned)(size_t)&kt[0][0];

  // Q fragments: rows q0+wave*16..+15, K = 128 channels in 4 steps of 32
  v16bf qf[4];
  {
    const __bf16* qrow =
        qT + ((size_t)b * NPIX + q0 + wave * 16 + l16) * C_INT + half * 8;
#pragma unroll
    for (int ks = 0; ks < 4; ++ks) qf[ks] = load_a16(qrow + ks * 32);
  }

  float mi[8], li[8];
  v8f   oacc[8];
#pragma unroll
  for (int r = 0; r < 8; ++r) { mi[r] = -1e30f; li[r] = 0.0f; }
#pragma unroll
  for (int ct = 0; ct < 8; ++ct)
#pragma unroll
    for (int r = 0; r < 8; ++r) oacc[ct][r] = 0.0f;

  for (int k0 = 0; k0 < NPIX; k0 += KT) {
    // ---- stage K tile via async global->LDS (16B/lane, 4 issues/thread) ----
    {
      const int seg = tid & 15;   // 16B segment within a 256B row
      for (int r = tid >> 4; r < KT; r += 16) {
        const __bf16* gp =
            kT + ((size_t)b * NPIX + k0 + r) * C_INT + seg * 8;
        async_copy16(kt_base + (unsigned)((r * LDC + seg * 8) * 2), gp);
      }
    }
    // ---- stage G tile transposed [c][key]; key-pairs packed into b32 stores
    for (int r = (tid >> 4) * 2; r < KT; r += 32) {
      const int cseg = (tid & 15) * 8;
      pk8 ga, gb;
      ga.u = *(const v4u*)(gT + ((size_t)b * NPIX + k0 + r) * C_INT + cseg);
      gb.u = *(const v4u*)(gT + ((size_t)b * NPIX + k0 + r + 1) * C_INT + cseg);
#pragma unroll
      for (int j = 0; j < 8; ++j) {
        pk2 p;
        p.h[0] = ga.h[j];
        p.h[1] = gb.h[j];
        *(unsigned*)&gt_t[cseg + j][r] = p.u;
      }
    }
    async_wait0();
    __syncthreads();

    // ---- S = Q K^T : 16 q-rows x 64 keys ----
    v8f sacc[4];
#pragma unroll
    for (int nt = 0; nt < 4; ++nt)
#pragma unroll
      for (int r = 0; r < 8; ++r) sacc[nt][r] = 0.0f;

#pragma unroll
    for (int ks = 0; ks < 4; ++ks) {
#pragma unroll
      for (int nt = 0; nt < 4; ++nt) {
        // logical B[c][key] stored kt[key][c] -> transposed contiguous load
        v16bf bf = load_bt16(&kt[nt * 16 + l16][ks * 32 + half * 8]);
        sacc[nt] = wmma_bf16(qf[ks], bf, sacc[nt]);
      }
    }

    // ---- online softmax: row = r + 8*half, 16 lanes per half hold cols ----
    float tmax[8];
#pragma unroll
    for (int r = 0; r < 8; ++r)
      tmax[r] = fmaxf(fmaxf(sacc[0][r], sacc[1][r]),
                      fmaxf(sacc[2][r], sacc[3][r]));
#pragma unroll
    for (int off = 1; off < 16; off <<= 1)
#pragma unroll
      for (int r = 0; r < 8; ++r)
        tmax[r] = fmaxf(tmax[r], __shfl_xor(tmax[r], off, 32));

    float scl[8], rsum[8];
#pragma unroll
    for (int r = 0; r < 8; ++r) {
      const float mn = fmaxf(mi[r], tmax[r]);
      scl[r]  = __expf(mi[r] - mn);
      mi[r]   = mn;
      rsum[r] = 0.0f;
    }
#pragma unroll
    for (int nt = 0; nt < 4; ++nt)
#pragma unroll
      for (int r = 0; r < 8; ++r) {
        const float p = __expf(sacc[nt][r] - mi[r]);
        sacc[nt][r]   = p;
        rsum[r] += p;
      }
#pragma unroll
    for (int off = 1; off < 16; off <<= 1)
#pragma unroll
      for (int r = 0; r < 8; ++r) rsum[r] += __shfl_xor(rsum[r], off, 32);
#pragma unroll
    for (int r = 0; r < 8; ++r) li[r] = li[r] * scl[r] + rsum[r];
#pragma unroll
    for (int ct = 0; ct < 8; ++ct)
#pragma unroll
      for (int r = 0; r < 8; ++r) oacc[ct][r] *= scl[r];

    // ---- P -> LDS (A-fragment layout source) ----
#pragma unroll
    for (int nt = 0; nt < 4; ++nt)
#pragma unroll
      for (int r = 0; r < 8; ++r)
        ps[wave][r + 8 * half][nt * 16 + l16] = (__bf16)sacc[nt][r];

    // ---- O += P * G ----
#pragma unroll
    for (int ks2 = 0; ks2 < 2; ++ks2) {
      v16bf pf = load_a16(&ps[wave][l16][ks2 * 32 + half * 8]);
#pragma unroll
      for (int ct = 0; ct < 8; ++ct) {
        // logical B[key][c] stored gt_t[c][key] -> transposed contiguous load
        v16bf gf = load_bt16(&gt_t[ct * 16 + l16][ks2 * 32 + half * 8]);
        oacc[ct] = wmma_bf16(pf, gf, oacc[ct]);
      }
    }
    __syncthreads();
  }

  // normalize + store y^T bf16
#pragma unroll
  for (int ct = 0; ct < 8; ++ct) {
    const int c = ct * 16 + l16;
#pragma unroll
    for (int r = 0; r < 8; ++r) {
      const int qn  = q0 + wave * 16 + r + 8 * half;
      const float o = oacc[ct][r] / li[r];
      yT[((size_t)b * NPIX + qn) * C_INT + c] = (__bf16)o;
    }
  }
}

// ---------------------------------------------------------------------------
// Kernel 3: out = w_w @ y + b_w + x   (fp32 output)
// ---------------------------------------------------------------------------
__global__ __launch_bounds__(256) void outproj_kernel(
    const __bf16* __restrict__ yT, const float* __restrict__ Ww,
    const float* __restrict__ bw, const float* __restrict__ x,
    float* __restrict__ out) {
  __shared__ __bf16 ys[128][LDC];  // [n][c], async-copied

  const int b    = blockIdx.y;
  const int col0 = blockIdx.x * 128;
  const int tid  = threadIdx.x;
  const int lane = tid & 31;
  const int wave = tid >> 5;
  const int half = lane >> 4;
  const int l16  = lane & 15;

  {
    const unsigned ys_base = (unsigned)(size_t)&ys[0][0];
    const int seg = tid & 15;
    for (int r = tid >> 4; r < 128; r += 16) {
      const __bf16* gp =
          yT + ((size_t)b * NPIX + col0 + r) * C_INT + seg * 8;
      async_copy16(ys_base + (unsigned)((r * LDC + seg * 8) * 2), gp);
    }
    async_wait0();
  }
  __syncthreads();

  for (int mt = 0; mt < 2; ++mt) {
    const int mbase = (wave * 2 + mt) * 16;
    v8f acc[8];
#pragma unroll
    for (int nt = 0; nt < 8; ++nt)
#pragma unroll
      for (int r = 0; r < 8; ++r) acc[nt][r] = 0.0f;

#pragma unroll
    for (int ks = 0; ks < 4; ++ks) {
      v16bf af =
          load_a_f32(Ww + (size_t)(mbase + l16) * C_INT + ks * 32 + half * 8);
#pragma unroll
      for (int nt = 0; nt < 8; ++nt) {
        // logical B[c][n] stored ys[n][c] -> transposed contiguous load
        v16bf bf = load_bt16(&ys[nt * 16 + l16][ks * 32 + half * 8]);
        acc[nt]  = wmma_bf16(af, bf, acc[nt]);
      }
    }

#pragma unroll
    for (int nt = 0; nt < 8; ++nt) {
      const int n = col0 + nt * 16 + l16;
#pragma unroll
      for (int r = 0; r < 8; ++r) {
        const int m    = mbase + r + 8 * half;
        const size_t i = ((size_t)b * C_IN + m) * NPIX + n;
        out[i] = acc[nt][r] + bw[m] + x[i];
      }
    }
  }
}

// ---------------------------------------------------------------------------
extern "C" void kernel_launch(void* const* d_in, const int* in_sizes, int n_in,
                              void* d_out, int out_size, void* d_ws,
                              size_t ws_size, hipStream_t stream) {
  (void)in_sizes; (void)n_in; (void)out_size; (void)ws_size;
  const float* x       = (const float*)d_in[0];
  const float* gamma   = (const float*)d_in[1];
  const float* beta    = (const float*)d_in[2];
  const float* mean    = (const float*)d_in[3];
  const float* var     = (const float*)d_in[4];
  const float* w_g     = (const float*)d_in[5];
  const float* b_g     = (const float*)d_in[6];
  const float* w_theta = (const float*)d_in[7];
  const float* b_theta = (const float*)d_in[8];
  const float* w_phi   = (const float*)d_in[9];
  const float* b_phi   = (const float*)d_in[10];
  const float* w_w     = (const float*)d_in[11];
  const float* b_w     = (const float*)d_in[12];

  const size_t seg = (size_t)BATCH * NPIX * C_INT;  // bf16 elements per tensor
  __bf16* gT = (__bf16*)d_ws;
  __bf16* qT = gT + seg;
  __bf16* kT = gT + 2 * seg;
  __bf16* yT = gT + 3 * seg;

  dim3 grid(NPIX / 128, BATCH);
  proj_kernel<<<grid, 256, 0, stream>>>(x, gamma, beta, mean, var, w_g, b_g, gT);
  proj_kernel<<<grid, 256, 0, stream>>>(x, gamma, beta, mean, var, w_theta,
                                        b_theta, qT);
  proj_kernel<<<grid, 256, 0, stream>>>(x, gamma, beta, mean, var, w_phi, b_phi,
                                        kT);
  attn_kernel<<<grid, 256, 0, stream>>>(qT, kT, gT, yT);
  outproj_kernel<<<grid, 256, 0, stream>>>(yT, w_w, b_w, x, (float*)d_out);
}
